// Attention_22110491640578
// MI455X (gfx1250) — matrix operand order repaired and checked by reading, not compiled, not run on hardware
//
#include <hip/hip_runtime.h>
#include <hip/hip_bf16.h>

// ---------------------------------------------------------------------------
// CDNA5 (gfx1250) attention block, bf16 WMMA path, vectorized fragment loads.
// All WMMA fragments are loaded as contiguous 128-bit vectors by pre-staging
// x (bf16), W^T (bf16) and V in a [B,H,dh,S] transposed bf16 layout.
// Layouts used by this code (wave32):
//   A-frag 16x32 : row = lane&15 ; elem e -> K = 16*hi + e
//   B-frag 32x16 : col = lane&15 ; elem e -> K = 16*hi + e   (contiguous in W^T)
//   C/D   16x16  : col = lane&15 ; elem v -> row = v + 8*hi
// ---------------------------------------------------------------------------

typedef __attribute__((ext_vector_type(16))) __bf16 v16bf;
typedef __attribute__((ext_vector_type(8)))  __bf16 v8bf;
typedef __attribute__((ext_vector_type(8)))  float  v8f;

constexpr int Bn = 4, Sn = 2048, Dn = 1024, Hn = 16, DH = 64;
constexpr int Mn = Bn * Sn;                 // 8192 rows of (b,s)

#define WMMA_BF16(A, B, C) \
    __builtin_amdgcn_wmma_f32_16x16x32_bf16(false, (A), false, (B), (short)0, (C), false, false)

__device__ __forceinline__ __bf16 f2bf(float f) { return (__bf16)f; }  // native v_cvt

__device__ __forceinline__ v16bf cat8(v8bf a, v8bf b) {
    return __builtin_shufflevector(a, b, 0,1,2,3,4,5,6,7,8,9,10,11,12,13,14,15);
}
__device__ __forceinline__ v16bf ld16(const __bf16* p) { return *(const v16bf*)p; }
__device__ __forceinline__ v8bf  ld8 (const __bf16* p) { return *(const v8bf*)p;  }

// ---------------------------------------------------------------------------
// One-shot staging kernels
// ---------------------------------------------------------------------------
__global__ __launch_bounds__(256)
void cvt_f32_to_bf16(const float* __restrict__ in, __bf16* __restrict__ out, int n)
{
    const int i = (blockIdx.x * 256 + threadIdx.x) * 8;
    if (i + 8 > n) return;
    v8bf r;
#pragma unroll
    for (int e = 0; e < 8; ++e) r[e] = f2bf(in[i + e]);
    *(v8bf*)(out + i) = r;
}

// out[Dn x Dn] = transpose(in[Dn x Dn]), fp32 -> bf16, 32x32 LDS tiles
__global__ __launch_bounds__(256)
void transpose_cvt(const float* __restrict__ in, __bf16* __restrict__ out)
{
    __shared__ __bf16 tile[32][33];
    const int tx = threadIdx.x & 31, ty = threadIdx.x >> 5;   // 32 x 8
    const int r0 = blockIdx.y * 32, c0 = blockIdx.x * 32;
#pragma unroll
    for (int i = 0; i < 32; i += 8)
        tile[ty + i][tx] = f2bf(in[(size_t)(r0 + ty + i) * Dn + c0 + tx]);
    __syncthreads();
#pragma unroll
    for (int i = 0; i < 32; i += 8)
        out[(size_t)(c0 + ty + i) * Dn + r0 + tx] = tile[tx][ty + i];
}

// ---------------------------------------------------------------------------
// GEMM: out[Mn,Dn] = A[Mn,Dn](bf16) * W(via W^T bf16) (+bias)
// block = 128 = 4 waves; wave tile = 32(M) x 64(N); 8 WMMA per K-step.
// Optional fp32 output; optional bf16 output (scaled), row-major or V^T-mode
// ([B,H,dh,S] layout for the attention V operand).
// ---------------------------------------------------------------------------
__global__ __launch_bounds__(128)
void gemm_wmma(const __bf16* __restrict__ A, const __bf16* __restrict__ WT,
               const float* __restrict__ bias, float* __restrict__ out_f32,
               __bf16* __restrict__ out_bf, float bf_scale, int bf_transposed)
{
    const int lane = threadIdx.x & 31, wave = threadIdx.x >> 5;
    const int lo = lane & 15, hi = lane >> 4;
    const int m0 = blockIdx.x * 32;
    const int n0 = blockIdx.y * 256 + wave * 64;

    const __bf16* a0 = A + (size_t)(m0 + lo) * Dn;
    const __bf16* a1 = A + (size_t)(m0 + 16 + lo) * Dn;

    v8f acc[2][4] = {};
    for (int kb = 0; kb < Dn; kb += 32) {
        const v16bf af0 = ld16(a0 + kb + 16 * hi);
        const v16bf af1 = ld16(a1 + kb + 16 * hi);
        if (kb + 32 < Dn) {
            __builtin_prefetch(a0 + kb + 32, 0, 1);
            __builtin_prefetch(WT + (size_t)(n0 + lo) * Dn + kb + 32, 0, 1);
        }
#pragma unroll
        for (int j = 0; j < 4; ++j) {
            const v16bf bf = ld16(WT + (size_t)(n0 + j * 16 + lo) * Dn + kb + 16 * hi);
            acc[0][j] = WMMA_BF16(af0, bf, acc[0][j]);
            acc[1][j] = WMMA_BF16(af1, bf, acc[1][j]);
        }
    }

#pragma unroll
    for (int j = 0; j < 4; ++j) {
        const int n = n0 + j * 16 + lo;
        const float bv = bias ? bias[n] : 0.0f;
#pragma unroll
        for (int mi = 0; mi < 2; ++mi) {
#pragma unroll
            for (int v = 0; v < 8; ++v) {
                const int m = m0 + mi * 16 + v + 8 * hi;
                const float val = acc[mi][j][v] + bv;
                if (out_f32) out_f32[(size_t)m * Dn + n] = val;
                if (out_bf) {
                    if (!bf_transposed) {
                        out_bf[(size_t)m * Dn + n] = f2bf(val * bf_scale);
                    } else {               // V^T: [B,H,dh,S]
                        const int b = m >> 11, s = m & (Sn - 1);   // Sn = 2048
                        const int h = n >> 6,  d = n & (DH - 1);   // dh = 64
                        out_bf[(((size_t)(b * Hn + h)) * DH + d) * Sn + s] =
                            f2bf(val * bf_scale);
                    }
                }
            }
        }
    }
}

// ---------------------------------------------------------------------------
// Causal flash attention. 128 threads = 4 independent waves, one 16-row
// q-tile each (per-wave LDS P buffer; no block-wide barrier — trip counts
// differ per wave). Q/K carry dh^-0.25 pre-scale. 32-key chunks:
//   2 S tiles (4 WMMA) -> fp32 online softmax (16-lane shfl_xor reductions)
//   -> P(16x32 bf16) via LDS (D-layout -> A-layout) -> 4 WMMA into O(16x64).
// ---------------------------------------------------------------------------
__global__ __launch_bounds__(128)
void flash_attn_wmma(const __bf16* __restrict__ Qb, const __bf16* __restrict__ Kb,
                     const __bf16* __restrict__ Vt, __bf16* __restrict__ Ah)
{
    __shared__ __bf16 Pl[4][16 * 32];

    const int lane = threadIdx.x & 31, wave = threadIdx.x >> 5;
    const int lo = lane & 15, hi = lane >> 4;
    const int qt = blockIdx.x * 4 + wave;
    const int h = blockIdx.y, b = blockIdx.z;
    const int qi0 = qt * 16, qend = qi0 + 15;
    __bf16* pl = Pl[wave];

    const size_t qk_base = (size_t)b * Sn * Dn + (size_t)h * DH;
    const __bf16* qp = Qb + qk_base;
    const __bf16* kp = Kb + qk_base;
    const __bf16* vp = Vt + ((size_t)(b * Hn + h)) * DH * Sn;   // [dh][S]

    v16bf qa0, qa1;
    {
        const __bf16* qr = qp + (size_t)(qi0 + lo) * Dn;
        qa0 = ld16(qr + 16 * hi);
        qa1 = ld16(qr + 32 + 16 * hi);
    }

    float mrow[8], lrow[8];
    v8f o[4] = {};
#pragma unroll
    for (int v = 0; v < 8; ++v) { mrow[v] = -1e30f; lrow[v] = 0.0f; }

    for (int kbase = 0; kbase <= qend; kbase += 32) {
        v8f s0 = {}, s1 = {};
        {
            const __bf16* kr = kp + (size_t)(kbase + lo) * Dn + 16 * hi;
            s0 = WMMA_BF16(qa0, ld16(kr), s0);
            s0 = WMMA_BF16(qa1, ld16(kr + 32), s0);
        }
        const bool t2 = (kbase + 16) <= qend;          // wave-uniform
        if (t2) {
            const __bf16* kr = kp + (size_t)(kbase + 16 + lo) * Dn + 16 * hi;
            s1 = WMMA_BF16(qa0, ld16(kr), s1);
            s1 = WMMA_BF16(qa1, ld16(kr + 32), s1);
        }

        // causal mask + fp32 online softmax; lane owns col lo, rows v+8*hi
#pragma unroll
        for (int v = 0; v < 8; ++v) {
            const int qi = qi0 + v + 8 * hi;
            if (kbase + lo > qi)             s0[v] = -1e9f;
            if (!t2 || kbase + 16 + lo > qi) s1[v] = -1e9f;

            float t = fmaxf(s0[v], s1[v]);
#pragma unroll
            for (int mk = 1; mk < 16; mk <<= 1) t = fmaxf(t, __shfl_xor(t, mk, 32));
            const float mn   = fmaxf(mrow[v], t);
            const float corr = __expf(mrow[v] - mn);
            const float p0   = __expf(s0[v] - mn);
            const float p1   = __expf(s1[v] - mn);
            float rs = p0 + p1;
#pragma unroll
            for (int mk = 1; mk < 16; mk <<= 1) rs += __shfl_xor(rs, mk, 32);
            lrow[v] = lrow[v] * corr + rs;
            mrow[v] = mn;
            o[0][v] *= corr; o[1][v] *= corr; o[2][v] *= corr; o[3][v] *= corr;

            pl[(v + 8 * hi) * 32 + lo]      = f2bf(p0);
            pl[(v + 8 * hi) * 32 + 16 + lo] = f2bf(p1);
        }

        // LDS hand-off: DS ops are in-order per wave; fence the compiler and
        // drain DScnt explicitly (CDNA5 split counter).
        asm volatile("s_wait_dscnt 0x0" ::: "memory");

        const v16bf pa = cat8(ld8(pl + lo * 32 + 16 * hi),
                              ld8(pl + lo * 32 + 16 * hi + 8));
#pragma unroll
        for (int dt = 0; dt < 4; ++dt) {
            const v16bf vf = ld16(vp + (size_t)(dt * 16 + lo) * Sn + kbase + 16 * hi);
            o[dt] = WMMA_BF16(pa, vf, o[dt]);
        }
    }

    // epilogue: O /= l, write merged-head bf16 activations [B,S,D]
    __bf16* ap = Ah + qk_base;
#pragma unroll
    for (int v = 0; v < 8; ++v) {
        const float inv = 1.0f / lrow[v];
        const size_t r = (size_t)(qi0 + v + 8 * hi) * Dn;
#pragma unroll
        for (int dt = 0; dt < 4; ++dt)
            ap[r + dt * 16 + lo] = f2bf(o[dt][v] * inv);
    }
}

// ---------------------------------------------------------------------------
extern "C" void kernel_launch(void* const* d_in, const int* in_sizes, int n_in,
                              void* d_out, int out_size, void* d_ws, size_t ws_size,
                              hipStream_t stream)
{
    const float* x  = (const float*)d_in[0];
    const float* Wq = (const float*)d_in[1];
    const float* bq = (const float*)d_in[2];
    const float* Wk = (const float*)d_in[3];
    const float* Wv = (const float*)d_in[4];
    const float* bv = (const float*)d_in[5];
    const float* Wc = (const float*)d_in[6];
    const float* bc = (const float*)d_in[7];
    // d_in[8] = n_head (16); dims fixed per reference.

    const float qk_scale = 0.35355339059327373f;   // 64^(-1/4)

    float* a_out = (float*)d_out;                  // [B,S,D]
    float* k_out = a_out + (size_t)Mn * Dn;        // hidden_state[0]
    float* v_out = k_out + (size_t)Mn * Dn;        // hidden_state[1]

    __bf16* xb  = (__bf16*)d_ws;                   // x, bf16           16 MB
    __bf16* WqT = xb  + (size_t)Mn * Dn;           // W^T bf16, 2 MB each
    __bf16* WkT = WqT + (size_t)Dn * Dn;
    __bf16* WvT = WkT + (size_t)Dn * Dn;
    __bf16* WcT = WvT + (size_t)Dn * Dn;
    __bf16* qb  = WcT + (size_t)Dn * Dn;           // q bf16 (scaled)   16 MB
    __bf16* kb  = qb  + (size_t)Mn * Dn;           // k bf16 (scaled)   16 MB
    __bf16* vt  = kb  + (size_t)Mn * Dn;           // V^T [B,H,dh,S]    16 MB
    __bf16* ah  = vt  + (size_t)Mn * Dn;           // attn out bf16     16 MB

    // ---- staging ----
    cvt_f32_to_bf16<<<(Mn * Dn) / (256 * 8), 256, 0, stream>>>(x, xb, Mn * Dn);
    dim3 tg(Dn / 32, Dn / 32), tb(256);
    transpose_cvt<<<tg, tb, 0, stream>>>(Wq, WqT);
    transpose_cvt<<<tg, tb, 0, stream>>>(Wk, WkT);
    transpose_cvt<<<tg, tb, 0, stream>>>(Wv, WvT);
    transpose_cvt<<<tg, tb, 0, stream>>>(Wc, WcT);

    // ---- projections ----
    dim3 gg(Mn / 32, Dn / 256), gb(128);
    gemm_wmma<<<gg, gb, 0, stream>>>(xb, WqT, bq, nullptr, qb, qk_scale, 0);
    gemm_wmma<<<gg, gb, 0, stream>>>(xb, WkT, nullptr, k_out, kb, qk_scale, 0);
    gemm_wmma<<<gg, gb, 0, stream>>>(xb, WvT, bv, v_out, vt, 1.0f, 1);

    // ---- causal flash attention ----
    flash_attn_wmma<<<dim3(Sn / 64, Hn, Bn), dim3(128), 0, stream>>>(qb, kb, vt, ah);

    // ---- output projection ----
    gemm_wmma<<<gg, gb, 0, stream>>>(ah, WcT, bc, a_out, nullptr, 1.0f, 0);
}
